// GlobalDecoder_43911745634362
// MI455X (gfx1250) — matrix operand order repaired and checked by reading, not compiled, not run on hardware
//
#include <hip/hip_runtime.h>

#define H 1024
#define NR 4096
#define NA 4096
#define NCL 8192
#define VOCAB 256

typedef __bf16 bf16_t;
typedef bf16_t v16bf __attribute__((ext_vector_type(16)));
typedef float  v8f   __attribute__((ext_vector_type(8)));

struct Q2 { uint4 a, b; };

static __device__ __forceinline__ unsigned short f2bf(float f) {
    unsigned u = __float_as_uint(f);
    unsigned r = (u + 0x7FFFu + ((u >> 16) & 1u)) >> 16;   // round-to-nearest-even
    return (unsigned short)r;
}
static __device__ __forceinline__ float bf2f(unsigned short s) {
    return __uint_as_float(((unsigned)s) << 16);
}
static __device__ __forceinline__ v16bf mkfrag(uint4 x, uint4 y) {
    Q2 q{x, y};
    return __builtin_bit_cast(v16bf, q);
}
// low 32 bits of a generic LDS address == wave-relative LDS offset (ISA 10.2)
static __device__ __forceinline__ unsigned lds_off(const void* p) {
    return (unsigned)(uintptr_t)p;
}

// ---------------------------------------------------------------- casts
__global__ void cast_f32_bf16(const float* __restrict__ in,
                              unsigned short* __restrict__ out, int n) {
    int i = blockIdx.x * blockDim.x + threadIdx.x;
    if (i < n) out[i] = f2bf(in[i]);
}

// ---------------------------------------------------------------- bf16 NT GEMM
// C[M,N] = A[M,K] * B[N,K]^T + bias (+relu), f32 accumulate via WMMA.
// M,N multiples of 128; K multiple of 32. Block: 256 thr (8 waves), tile 128x128.
// BK=32, double-buffered LDS (40 KB), async global->LDS DMA (ASYNCcnt) software
// pipeline: stage k+1 issued right after the barrier, overlapped with WMMAs of
// stage k. One barrier + one s_wait_asynccnt per K-step.
template <int RELU, int WF, int WBF>
__global__ __launch_bounds__(256) void gemm_nt_bf16(
    const unsigned short* __restrict__ A,
    const unsigned short* __restrict__ B,
    const float* __restrict__ bias,
    float* __restrict__ outF,
    unsigned short* __restrict__ outBF,
    int M, int Nn, int K)
{
    constexpr int LDT = 40;                       // 80B row stride: conflict-free b128
    __shared__ unsigned short sA[2][128 * LDT];
    __shared__ unsigned short sB[2][128 * LDT];

    const int tid   = threadIdx.x;
    const int lane  = tid & 31;
    const int wid   = tid >> 5;
    const int waveM = wid & 1;                    // 2 waves along M
    const int waveN = wid >> 1;                   // 4 waves along N
    const int bm    = blockIdx.y * 128;
    const int bn    = blockIdx.x * 128;

    v8f acc[4][2];
#pragma unroll
    for (int i = 0; i < 4; i++)
#pragma unroll
        for (int j = 0; j < 2; j++) acc[i][j] = v8f{0,0,0,0,0,0,0,0};

    const int grow = tid >> 2;                    // 0..63
    const int gseg = (tid & 3) * 8;               // ushort offset in 32-wide K tile
    const int lm = lane & 15, lh = lane >> 4;

    // issue one stage of async global->LDS copies (4 x b128 per thread)
    auto stage = [&](int buf, int k0) {
        const unsigned short* ga0 = A + (size_t)(bm + grow) * K + k0 + gseg;
        const unsigned short* ga1 = A + (size_t)(bm + grow + 64) * K + k0 + gseg;
        const unsigned short* gb0 = B + (size_t)(bn + grow) * K + k0 + gseg;
        const unsigned short* gb1 = B + (size_t)(bn + grow + 64) * K + k0 + gseg;
        unsigned la0 = lds_off(&sA[buf][grow * LDT + gseg]);
        unsigned la1 = lds_off(&sA[buf][(grow + 64) * LDT + gseg]);
        unsigned lb0 = lds_off(&sB[buf][grow * LDT + gseg]);
        unsigned lb1 = lds_off(&sB[buf][(grow + 64) * LDT + gseg]);
        asm volatile(
            "global_load_async_to_lds_b128 %0, %4, off\n\t"
            "global_load_async_to_lds_b128 %1, %5, off\n\t"
            "global_load_async_to_lds_b128 %2, %6, off\n\t"
            "global_load_async_to_lds_b128 %3, %7, off"
            :: "v"(la0), "v"(la1), "v"(lb0), "v"(lb1),
               "v"(ga0), "v"(ga1), "v"(gb0), "v"(gb1)
            : "memory");
    };

    stage(0, 0);
    int buf = 0;
    for (int k0 = 0; k0 < K; k0 += 32, buf ^= 1) {
        asm volatile("s_wait_asynccnt 0x0" ::: "memory");
        __syncthreads();
        if (k0 + 32 < K) stage(buf ^ 1, k0 + 32);   // overlap with WMMAs below

        // B frag: lane=N col, 16 contiguous K (lanes 0-15: K0-15, 16-31: K16-31)
        v16bf bfrag[2];
#pragma unroll
        for (int j = 0; j < 2; j++) {
            const unsigned short* p =
                &sB[buf][(waveN * 32 + j * 16 + lm) * LDT + lh * 16];
            bfrag[j] = mkfrag(*(const uint4*)p, *(const uint4*)(p + 8));
        }
        // A frag: lane=M row, 16 contiguous K (lanes 0-15: K0-15, 16-31: K16-31)
#pragma unroll
        for (int i = 0; i < 4; i++) {
            const unsigned short* p =
                &sA[buf][(waveM * 64 + i * 16 + lm) * LDT + lh * 16];
            v16bf afrag = mkfrag(*(const uint4*)p, *(const uint4*)(p + 8));
#pragma unroll
            for (int j = 0; j < 2; j++) {
                acc[i][j] = __builtin_amdgcn_wmma_f32_16x16x32_bf16(
                    false, afrag, false, bfrag[j], (short)0, acc[i][j],
                    false, false);
            }
        }
    }

    // epilogue: C VGPR v -> (M = i*16 + v + 8*(lane>>4), N = lane&15); branch-free
#pragma unroll
    for (int j = 0; j < 2; j++) {
        const int col = bn + waveN * 32 + j * 16 + lm;
        const float bv = bias[col];
#pragma unroll
        for (int i = 0; i < 4; i++) {
            const int row0 = bm + waveM * 64 + i * 16 + lh * 8;
#pragma unroll
            for (int v = 0; v < 8; v++) {
                float val = acc[i][j][v] + bv;
                if (RELU) val = fmaxf(val, 0.0f);
                const size_t o = (size_t)(row0 + v) * Nn + col;
                if (WF)  outF[o]  = val;
                if (WBF) outBF[o] = f2bf(val);
            }
        }
    }
}

// ---------------------------------------------------------------- gh0 = W_hh @ state + b_hh
__global__ __launch_bounds__(256) void gemv_gh0(
    const float* __restrict__ Whh, const float* __restrict__ state,
    const float* __restrict__ bhh, float* __restrict__ gh0)
{
    const int row  = blockIdx.x * 8 + (threadIdx.x >> 5);
    const int lane = threadIdx.x & 31;
    const float* wr = Whh + (size_t)row * H;
    float s = 0.f;
    for (int k = lane; k < H; k += 32) s += wr[k] * state[k];
#pragma unroll
    for (int off = 16; off; off >>= 1) s += __shfl_xor(s, off, 32);
    if (lane == 0) gh0[row] = s + bhh[row];
}

// ---------------------------------------------------------------- GRU combine
static __device__ __forceinline__ float gru_out(float ir, float iz, float in_,
                                                float hr, float hz, float hn, float h) {
    float r = 1.f / (1.f + __expf(-(ir + hr)));
    float z = 1.f / (1.f + __expf(-(iz + hz)));
    float n = tanhf(in_ + r * hn);
    return (1.f - z) * n + z * h;
}

// h1[v] for all 256 vocab entries (h = state broadcast)
__global__ void gru_step1(const float* __restrict__ GI, const float* __restrict__ gh0,
                          const float* __restrict__ state,
                          float* __restrict__ h1f, unsigned short* __restrict__ h1b)
{
    int idx = blockIdx.x * blockDim.x + threadIdx.x;       // VOCAB*H
    int v = idx >> 10, j = idx & (H - 1);
    const float* gi = GI + (size_t)v * 3 * H;
    float o = gru_out(gi[j], gi[H + j], gi[2 * H + j],
                      gh0[j], gh0[H + j], gh0[2 * H + j], state[j]);
    h1f[idx] = o;
    h1b[idx] = f2bf(o);
}

// step 2 for all 8192 clauses: h=h1[tok0] (table), gh=GH1[tok0] (table), gi=GI[tok1]
__global__ void gru_step2(const float* __restrict__ GI, const float* __restrict__ GH1,
                          const float* __restrict__ h1f,
                          const int* __restrict__ rel, const int* __restrict__ attr,
                          float* __restrict__ h2f, unsigned short* __restrict__ h2b,
                          unsigned short* __restrict__ clause_bf)
{
    int idx = blockIdx.x * blockDim.x + threadIdx.x;       // NCL*H
    int row = idx >> 10, j = idx & (H - 1);
    int t0, t1;
    if (row < NR) { t0 = rel[row * 3]; t1 = rel[row * 3 + 1]; }
    else          { int a = row - NR; t0 = attr[a * 2]; t1 = attr[a * 2 + 1]; }
    const float* gi = GI  + (size_t)t1 * 3 * H;
    const float* gh = GH1 + (size_t)t0 * 3 * H;
    float h = h1f[(size_t)t0 * H + j];
    float o = gru_out(gi[j], gi[H + j], gi[2 * H + j],
                      gh[j], gh[H + j], gh[2 * H + j], h);
    if (row < NR) {                      // rel: one more step pending
        h2f[(size_t)row * H + j] = o;
        h2b[(size_t)row * H + j] = f2bf(o);
    } else {                             // attr: final hidden -> clause matrix
        clause_bf[(size_t)row * H + j] = f2bf(o);
    }
}

// step 3 (rel only): gh=GH2[row] (full GEMM result), gi=GI[tok2]
__global__ void gru_step3(const float* __restrict__ GI, const float* __restrict__ GH2,
                          const float* __restrict__ h2f, const int* __restrict__ rel,
                          unsigned short* __restrict__ clause_bf)
{
    int idx = blockIdx.x * blockDim.x + threadIdx.x;       // NR*H
    int row = idx >> 10, j = idx & (H - 1);
    int t2 = rel[row * 3 + 2];
    const float* gi = GI  + (size_t)t2 * 3 * H;
    const float* gh = GH2 + (size_t)row * 3 * H;
    float o = gru_out(gi[j], gi[H + j], gi[2 * H + j],
                      gh[j], gh[H + j], gh[2 * H + j], h2f[(size_t)row * H + j]);
    clause_bf[(size_t)row * H + j] = f2bf(o);
}

// ---------------------------------------------------------------- logits GEMV
__global__ __launch_bounds__(256) void gemv_logits(
    const unsigned short* __restrict__ X2, const float* __restrict__ W3,
    const float* __restrict__ b3, float* __restrict__ logits)
{
    const int row  = blockIdx.x * 8 + (threadIdx.x >> 5);
    const int lane = threadIdx.x & 31;
    const unsigned short* xr = X2 + (size_t)row * H;
    float s = 0.f;
    for (int k = lane; k < H; k += 32) s += bf2f(xr[k]) * W3[k];
#pragma unroll
    for (int off = 16; off; off >>= 1) s += __shfl_xor(s, off, 32);
    if (lane == 0) logits[row] = s + b3[0];
}

// ---------------------------------------------------------------- softmax + eps smoothing
__global__ __launch_bounds__(1024) void softmax_eps(
    const float* __restrict__ logits, const float* __restrict__ eps,
    float* __restrict__ out)
{
    __shared__ float red[1024];
    const int t = threadIdx.x;
    float l[8], mx = -1e30f;
#pragma unroll
    for (int i = 0; i < 8; i++) { l[i] = logits[t + i * 1024]; mx = fmaxf(mx, l[i]); }
    red[t] = mx; __syncthreads();
    for (int s = 512; s; s >>= 1) { if (t < s) red[t] = fmaxf(red[t], red[t + s]); __syncthreads(); }
    mx = red[0]; __syncthreads();
    float sum = 0.f;
#pragma unroll
    for (int i = 0; i < 8; i++) { l[i] = __expf(l[i] - mx); sum += l[i]; }
    red[t] = sum; __syncthreads();
    for (int s = 512; s; s >>= 1) { if (t < s) red[t] += red[t + s]; __syncthreads(); }
    const float inv = 1.0f / red[0];
    const float e = eps[0];
#pragma unroll
    for (int i = 0; i < 8; i++)
        out[t + i * 1024] = l[i] * inv * (1.f - e) + e / (float)NCL;
}

// ================================================================ launch
extern "C" void kernel_launch(void* const* d_in, const int* in_sizes, int n_in,
                              void* d_out, int out_size, void* d_ws, size_t ws_size,
                              hipStream_t stream) {
    const float* state = (const float*)d_in[0];
    const int*   rel   = (const int*)d_in[1];
    const int*   attr  = (const int*)d_in[2];
    const float* table = (const float*)d_in[3];
    const float* W_emb = (const float*)d_in[4];
    const float* b_emb = (const float*)d_in[5];
    const float* W_ih  = (const float*)d_in[6];
    const float* W_hh  = (const float*)d_in[7];
    const float* b_ih  = (const float*)d_in[8];
    const float* b_hh  = (const float*)d_in[9];
    const float* W1    = (const float*)d_in[10];
    const float* b1    = (const float*)d_in[11];
    const float* W2    = (const float*)d_in[12];
    const float* b2    = (const float*)d_in[13];
    const float* W3    = (const float*)d_in[14];
    const float* b3    = (const float*)d_in[15];
    const float* eps   = (const float*)d_in[16];

    char* ws = (char*)d_ws;
    size_t off = 0;
    auto alloc = [&](size_t bytes) -> void* {
        void* p = ws + off;
        off = (off + bytes + 255) & ~(size_t)255;
        return p;
    };

    unsigned short* tb_bf    = (unsigned short*)alloc((size_t)VOCAB * H * 2);
    unsigned short* Wemb_bf  = (unsigned short*)alloc((size_t)H * H * 2);
    unsigned short* Wih_bf   = (unsigned short*)alloc((size_t)3 * H * H * 2);
    unsigned short* Whh_bf   = (unsigned short*)alloc((size_t)3 * H * H * 2);
    unsigned short* W1_bf    = (unsigned short*)alloc((size_t)H * H * 2);
    unsigned short* W2_bf    = (unsigned short*)alloc((size_t)H * H * 2);
    unsigned short* P_bf     = (unsigned short*)alloc((size_t)VOCAB * H * 2);
    float*          GI       = (float*)alloc((size_t)VOCAB * 3 * H * 4);
    float*          gh0      = (float*)alloc((size_t)3 * H * 4);
    float*          h1f      = (float*)alloc((size_t)VOCAB * H * 4);
    unsigned short* h1b      = (unsigned short*)alloc((size_t)VOCAB * H * 2);
    float*          GH1      = (float*)alloc((size_t)VOCAB * 3 * H * 4);
    float*          h2f      = (float*)alloc((size_t)NR * H * 4);
    unsigned short* h2b      = (unsigned short*)alloc((size_t)NR * H * 2);
    float*          GH2      = (float*)alloc((size_t)NR * 3 * H * 4);
    unsigned short* clausebf = (unsigned short*)alloc((size_t)NCL * H * 2);
    unsigned short* X1_bf    = (unsigned short*)alloc((size_t)NCL * H * 2);
    unsigned short* X2_bf    = (unsigned short*)alloc((size_t)NCL * H * 2);
    float*          logits   = (float*)alloc((size_t)NCL * 4);

    auto cast = [&](const float* src, unsigned short* dst, int n) {
        cast_f32_bf16<<<(n + 255) / 256, 256, 0, stream>>>(src, dst, n);
    };
    cast(table, tb_bf,   VOCAB * H);
    cast(W_emb, Wemb_bf, H * H);
    cast(W_ih,  Wih_bf,  3 * H * H);
    cast(W_hh,  Whh_bf,  3 * H * H);
    cast(W1,    W1_bf,   H * H);
    cast(W2,    W2_bf,   H * H);

    // P = table @ W_emb^T + b_emb          [256,1024]  (bf16 only)
    gemm_nt_bf16<0, 0, 1><<<dim3(H / 128, VOCAB / 128), 256, 0, stream>>>(
        tb_bf, Wemb_bf, b_emb, nullptr, P_bf, VOCAB, H, H);
    // GI = P @ W_ih^T + b_ih               [256,3072]  (f32, gathered later)
    gemm_nt_bf16<0, 1, 0><<<dim3(3 * H / 128, VOCAB / 128), 256, 0, stream>>>(
        P_bf, Wih_bf, b_ih, GI, nullptr, VOCAB, 3 * H, H);
    // gh0 = W_hh @ state + b_hh            [3072]
    gemv_gh0<<<(3 * H) / 8, 256, 0, stream>>>(W_hh, state, b_hh, gh0);
    // h1 table                              [256,1024]
    gru_step1<<<(VOCAB * H) / 256, 256, 0, stream>>>(GI, gh0, state, h1f, h1b);
    // GH1 = h1 @ W_hh^T + b_hh             [256,3072]
    gemm_nt_bf16<0, 1, 0><<<dim3(3 * H / 128, VOCAB / 128), 256, 0, stream>>>(
        h1b, Whh_bf, b_hh, GH1, nullptr, VOCAB, 3 * H, H);
    // step 2: all clauses (attr finishes here)
    gru_step2<<<(NCL * H) / 256, 256, 0, stream>>>(
        GI, GH1, h1f, rel, attr, h2f, h2b, clausebf);
    // GH2 = h2_rel @ W_hh^T + b_hh         [4096,3072]  (big GEMM)
    gemm_nt_bf16<0, 1, 0><<<dim3(3 * H / 128, NR / 128), 256, 0, stream>>>(
        h2b, Whh_bf, b_hh, GH2, nullptr, NR, 3 * H, H);
    // step 3: rel clauses finish
    gru_step3<<<(NR * H) / 256, 256, 0, stream>>>(GI, GH2, h2f, rel, clausebf);
    // MLP
    gemm_nt_bf16<1, 0, 1><<<dim3(H / 128, NCL / 128), 256, 0, stream>>>(
        clausebf, W1_bf, b1, nullptr, X1_bf, NCL, H, H);
    gemm_nt_bf16<1, 0, 1><<<dim3(H / 128, NCL / 128), 256, 0, stream>>>(
        X1_bf, W2_bf, b2, nullptr, X2_bf, NCL, H, H);
    // logits + softmax(+eps)
    gemv_logits<<<NCL / 8, 256, 0, stream>>>(X2_bf, W3, b3, logits);
    softmax_eps<<<1, 1024, 0, stream>>>(logits, eps, (float*)d_out);
}
